// TokenAggregationforCap_69595650064757
// MI455X (gfx1250) — matrix operand order repaired and checked by reading, not compiled, not run on hardware
//
#include <hip/hip_runtime.h>
#include <hip/hip_bf16.h>

typedef __attribute__((ext_vector_type(16))) __bf16        v16bf;
typedef __attribute__((ext_vector_type(8)))  float         v8f;
typedef __attribute__((ext_vector_type(4)))  unsigned int  v4u;

union ABfrag { v16bf v; unsigned int u[8]; v4u q[2]; };

#define DIM    512
#define HIDDEN 102
#define HPAD   128
#define KTOK   64
#define NTOK   65536
#define NPAIR  (NTOK / 2)     // 32768 token pairs
#define LN_EPS 1e-5f
#define NBLK1  512            // score kernel workgroups (128 tokens each)
#define NCHUNK 64             // aggregation workgroups (1024 tokens each)
#define NEBLK  512            // epack workgroups (128 tokens each)

// ---- workspace layout (dword offsets) -------------------------------------
#define WS_SBUF     0                                   // [token][64] scores
#define WS_BLOCKMAX (NTOK * KTOK)                       // 512 x 64
#define WS_KMAX     (WS_BLOCKMAX + NBLK1 * KTOK)        // 64
#define WS_SUMEXP   (WS_KMAX + KTOK)                    // 64
#define WS_ACCUM    (WS_SUMEXP + KTOK)                  // 64 x 512
#define WS_ESUMP    (WS_ACCUM + KTOK * DIM)             // 512 x 64
#define WS_W1P      (WS_ESUMP + NEBLK * KTOK)           // [n=128][kp=256] uint
#define WS_W2P      (WS_W1P + HPAD * (DIM / 2))         // [n=64][hp=64] uint
#define WS_EBUF     (WS_W2P + KTOK * (HPAD / 2))        // [k=64][np=32768] uint
#define WS_XP       (WS_EBUF + KTOK * NPAIR)            // [d=512][np=32768] uint
#define WS_END_XP   (WS_XP + DIM * NPAIR)

#define NEG_BIG (-3.402823466e38f)

__device__ __forceinline__ int koffA(int v) { return (v < 4) ? 2 * v : 16 + 2 * (v - 4); }

__device__ __forceinline__ unsigned int pack2bf(float f0, float f1)
{
    union { __bf16 b; unsigned short s; } c0, c1;
    c0.b = (__bf16)f0;
    c1.b = (__bf16)f1;
    return (unsigned int)c0.s | ((unsigned int)c1.s << 16);
}

// ---------------------------------------------------------------------------
// Kernel 0a: pack W1 / W2 -> bf16 pair dwords, COLUMN-major (n-major):
//   w1p2[n][kp] = bf16(W1[2kp][n]) | bf16(W1[2kp+1][n])<<16
//   w2p2[n][hp] = bf16(W2[2hp][n]) | bf16(W2[2hp+1][n])<<16
// ---------------------------------------------------------------------------
__global__ void pack_weights_kernel(const float* __restrict__ W1,
                                    const float* __restrict__ W2,
                                    float* __restrict__ ws)
{
    unsigned int* w1p = (unsigned int*)(ws + WS_W1P);
    unsigned int* w2p = (unsigned int*)(ws + WS_W2P);
    int i = blockIdx.x * blockDim.x + threadIdx.x;
    const int n1 = HPAD * (DIM / 2);           // 32768
    const int n2 = KTOK * (HPAD / 2);          // 4096
    if (i < n1) {
        int n = i >> 8, kp = i & 255;
        float f0 = (n < HIDDEN) ? W1[(size_t)(2 * kp) * HIDDEN + n] : 0.f;
        float f1 = (n < HIDDEN) ? W1[(size_t)(2 * kp + 1) * HIDDEN + n] : 0.f;
        w1p[i] = pack2bf(f0, f1);
    } else if (i < n1 + n2) {
        int j = i - n1;
        int n = j >> 6, hp = j & 63;
        float f0 = (2 * hp     < HIDDEN) ? W2[(2 * hp) * KTOK + n] : 0.f;
        float f1 = (2 * hp + 1 < HIDDEN) ? W2[(2 * hp + 1) * KTOK + n] : 0.f;
        w2p[j] = pack2bf(f0, f1);
    }
}

// ---------------------------------------------------------------------------
// Kernel 0b (optional): x -> bf16 token-pair dwords, d-major: xp2[d][np]
// LDS-tiled 64x64 transpose: coalesced reads and writes.
// ---------------------------------------------------------------------------
__global__ __launch_bounds__(256) void pack_x_kernel(const float* __restrict__ x,
                                                     float* __restrict__ ws)
{
    __shared__ float tile[64][65];
    unsigned int* xp = (unsigned int*)(ws + WS_XP);
    const int tok0 = (blockIdx.x >> 3) * 64;
    const int d0   = (blockIdx.x & 7) * 64;
    // coalesced load of [64 tokens][64 dims]
    for (int it = 0; it < 16; ++it) {
        int idx = it * 256 + threadIdx.x;
        int row = idx >> 6, col = idx & 63;
        tile[row][col] = x[(size_t)(tok0 + row) * DIM + d0 + col];
    }
    __syncthreads();
    // coalesced packed-transposed store
    const int dl  = threadIdx.x >> 2;
    const int grp = threadIdx.x & 3;
    #pragma unroll
    for (int j = 0; j < 8; ++j) {
        int npl = grp * 8 + j;                     // 0..31
        unsigned int p = pack2bf(tile[2 * npl][dl], tile[2 * npl + 1][dl]);
        xp[(size_t)(d0 + dl) * NPAIR + (tok0 >> 1) + npl] = p;
    }
}

// ---------------------------------------------------------------------------
// Kernel 1: LayerNorm + fc1(WMMA bf16) + GELU + fc2(WMMA bf16) -> scores
// ---------------------------------------------------------------------------
__global__ __launch_bounds__(256) void score_kernel(
    const float* __restrict__ x, const float* __restrict__ gamma,
    const float* __restrict__ beta, const float* __restrict__ b1,
    const float* __restrict__ b2, const float* __restrict__ scale,
    float* __restrict__ ws)
{
    __shared__ __bf16 lds_h[8][16 * HPAD];   // per-wave GELU(hidden) tile, 32 KB
    __shared__ float  lds_max[8][KTOK];
    __shared__ float  lds_g[DIM], lds_b[DIM];

    const unsigned int* w1p = (const unsigned int*)(ws + WS_W1P);
    const unsigned int* w2p = (const unsigned int*)(ws + WS_W2P);

    for (int k = threadIdx.x; k < DIM; k += 256) {
        lds_g[k] = gamma[k];
        lds_b[k] = beta[k];
    }

    const int wave = threadIdx.x >> 5;
    const int lane = threadIdx.x & 31;
    const int l15  = lane & 15;
    const int hi   = lane >> 4;
    const int tile = blockIdx.x * 8 + wave;
    const int row0 = tile * 16;

    // ---- LayerNorm stats: lanes (l15, l15+16) each sum half of row l15
    const float* xr = x + (size_t)(row0 + l15) * DIM;
    float s = 0.f, sq = 0.f;
    for (int k = hi * 256; k < hi * 256 + 256; k += 4) {
        float4 v = *(const float4*)(xr + k);
        s  += v.x + v.y + v.z + v.w;
        sq += v.x * v.x + v.y * v.y + v.z * v.z + v.w * v.w;
    }
    s  += __shfl_xor(s, 16, 32);
    sq += __shfl_xor(sq, 16, 32);
    const float mu   = s * (1.f / DIM);
    const float rstd = rsqrtf(sq * (1.f / DIM) - mu * mu + LN_EPS);
    __syncthreads();

    // ---- fc1: [16 x 512] @ [512 x HPAD]
    v8f acc1[8] = {};
    for (int kk0 = 0; kk0 < DIM; kk0 += 32) {
        ABfrag a;
        #pragma unroll
        for (int v = 0; v < 8; ++v) {
            int kb = kk0 + 2 * v + hi * 16;              // even
            float2 xv = *(const float2*)(xr + kb);
            float2 gv = *(const float2*)(lds_g + kb);
            float2 bv = *(const float2*)(lds_b + kb);
            a.u[v] = pack2bf((xv.x - mu) * rstd * gv.x + bv.x,
                             (xv.y - mu) * rstd * gv.y + bv.y);
        }
        #pragma unroll
        for (int t = 0; t < 8; ++t) {
            int n = t * 16 + l15;
            const unsigned int* bb = w1p + (size_t)n * (DIM / 2) + (kk0 >> 1) + hi * 8;
            ABfrag b;
            b.q[0] = *(const v4u*)(bb);
            b.q[1] = *(const v4u*)(bb + 4);
            acc1[t] = __builtin_amdgcn_wmma_f32_16x16x32_bf16(
                false, a.v, false, b.v, (short)0, acc1[t], false, false);
        }
    }

    // ---- bias + exact GELU -> LDS bf16 tile [m][h]
    #pragma unroll
    for (int t = 0; t < 8; ++t) {
        int n = t * 16 + l15;
        float bias = (n < HIDDEN) ? b1[n] : 0.f;
        #pragma unroll
        for (int r = 0; r < 8; ++r) {
            int m = r + 8 * hi;
            float v = acc1[t][r] + bias;
            float g = 0.5f * v * (1.f + erff(v * 0.70710678118f));
            if (n >= HIDDEN) g = 0.f;
            lds_h[wave][m * HPAD + n] = (__bf16)g;
        }
    }
    __syncthreads();

    // ---- fc2: [16 x HPAD] @ [HPAD x 64]
    const unsigned int* hrow = (const unsigned int*)&lds_h[wave][l15 * HPAD];
    v8f acc2[4] = {};
    for (int kk0 = 0; kk0 < HPAD; kk0 += 32) {
        int dwbase = (kk0 >> 1) + hi * 8;                // A-layout pair dwords
        ABfrag a;
        a.q[0] = *(const v4u*)(hrow + dwbase);
        a.q[1] = *(const v4u*)(hrow + dwbase + 4);
        #pragma unroll
        for (int t = 0; t < 4; ++t) {
            int n = t * 16 + l15;
            const unsigned int* bb = w2p + (size_t)n * (HPAD / 2) + (kk0 >> 1) + hi * 8;
            ABfrag b;
            b.q[0] = *(const v4u*)(bb);
            b.q[1] = *(const v4u*)(bb + 4);
            acc2[t] = __builtin_amdgcn_wmma_f32_16x16x32_bf16(
                false, a.v, false, b.v, (short)0, acc2[t], false, false);
        }
    }

    // ---- scores = (acc + b2) * scale -> sbuf[token][64]; block max
    const float sc = scale[0];
    float* sbuf = ws + WS_SBUF;
    float mx[4];
    #pragma unroll
    for (int t = 0; t < 4; ++t) {
        int n = t * 16 + l15;
        float bias = b2[n];
        float m_t = NEG_BIG;
        #pragma unroll
        for (int r = 0; r < 8; ++r) {
            int tok = row0 + r + 8 * hi;
            float sv = (acc2[t][r] + bias) * sc;
            sbuf[(size_t)tok * KTOK + n] = sv;
            m_t = fmaxf(m_t, sv);
        }
        mx[t] = m_t;
    }
    #pragma unroll
    for (int t = 0; t < 4; ++t) {
        float m2 = fmaxf(mx[t], __shfl_xor(mx[t], 16, 32));
        if (hi == 0) lds_max[wave][t * 16 + l15] = m2;
    }
    __syncthreads();
    if (threadIdx.x < KTOK) {
        float m = lds_max[0][threadIdx.x];
        #pragma unroll
        for (int w = 1; w < 8; ++w) m = fmaxf(m, lds_max[w][threadIdx.x]);
        ws[WS_BLOCKMAX + (size_t)blockIdx.x * KTOK + threadIdx.x] = m;
    }
}

// ---------------------------------------------------------------------------
// Kernel 2: reduce per-block maxima -> kmax[64]; zero accumulators
// ---------------------------------------------------------------------------
__global__ void reduce_init_kernel(float* __restrict__ ws)
{
    int tid = threadIdx.x;
    if (tid < KTOK) {
        float m = NEG_BIG;
        for (int b = 0; b < NBLK1; ++b)
            m = fmaxf(m, ws[WS_BLOCKMAX + (size_t)b * KTOK + tid]);
        ws[WS_KMAX + tid] = m;
    }
    for (int i = tid; i < KTOK * DIM; i += blockDim.x)
        ws[WS_ACCUM + i] = 0.f;
}

// ---------------------------------------------------------------------------
// Kernel 3: e = exp(s - kmax) -> k-major bf16 pair dwords ep2[k][np]
// LDS-tiled transpose (128 tokens per block) + per-block sum partials.
// ---------------------------------------------------------------------------
__global__ __launch_bounds__(256) void epack_kernel(float* __restrict__ ws)
{
    __shared__ float tile[128][65];
    __shared__ float lse[256];
    unsigned int* ep = (unsigned int*)(ws + WS_EBUF);
    const float* sbuf = ws + WS_SBUF;
    const int tok0 = blockIdx.x * 128;
    // coalesced load of [128 tokens][64 k]
    for (int it = 0; it < 32; ++it) {
        int idx = it * 256 + threadIdx.x;
        int row = idx >> 6, col = idx & 63;
        tile[row][col] = sbuf[(size_t)(tok0 + row) * KTOK + col];
    }
    __syncthreads();
    const int k   = threadIdx.x >> 2;
    const int grp = threadIdx.x & 3;
    const float kmaxv = ws[WS_KMAX + k];
    float esum = 0.f;
    #pragma unroll
    for (int j = 0; j < 16; ++j) {
        int npl = grp * 16 + j;                    // 0..63
        float e0 = __expf(tile[2 * npl][k]     - kmaxv);
        float e1 = __expf(tile[2 * npl + 1][k] - kmaxv);
        ep[(size_t)k * NPAIR + (tok0 >> 1) + npl] = pack2bf(e0, e1);
        esum += e0 + e1;
    }
    lse[threadIdx.x] = esum;
    __syncthreads();
    if (threadIdx.x < KTOK)
        ws[WS_ESUMP + (size_t)blockIdx.x * KTOK + threadIdx.x] =
            lse[threadIdx.x * 4] + lse[threadIdx.x * 4 + 1] +
            lse[threadIdx.x * 4 + 2] + lse[threadIdx.x * 4 + 3];
}

// ---------------------------------------------------------------------------
// Kernel 4: reduce block sum partials -> sumexp[64]
// ---------------------------------------------------------------------------
__global__ void reduce_sum_kernel(float* __restrict__ ws)
{
    int tid = threadIdx.x;
    if (tid < KTOK) {
        float s = 0.f;
        for (int b = 0; b < NEBLK; ++b)
            s += ws[WS_ESUMP + (size_t)b * KTOK + tid];
        ws[WS_SUMEXP + tid] = s;
    }
}

// ---------------------------------------------------------------------------
// Kernel 5: out_unnorm += e @ x  (bf16 WMMA, f32 accum).
// A: two b128 loads from k-major ep2. B (PACKED): two b128 from d-major xp2.
// ---------------------------------------------------------------------------
template <bool PACKED>
__global__ __launch_bounds__(256) void agg_kernel(
    const float* __restrict__ x, float* __restrict__ ws)
{
    const int wave = threadIdx.x >> 5;
    const int lane = threadIdx.x & 31;
    const int l15  = lane & 15;
    const int hi   = lane >> 4;
    const int m4   = wave & 3;                  // which 16 k-rows
    const int dh   = wave >> 2;                 // which 256-wide d half
    const int kk   = m4 * 16 + l15;
    const unsigned int* ep = (const unsigned int*)(ws + WS_EBUF) + (size_t)kk * NPAIR;
    const unsigned int* xp = (const unsigned int*)(ws + WS_XP);
    const int n_start = blockIdx.x * (NTOK / NCHUNK);

    v8f acc[16] = {};

    for (int n0 = n_start; n0 < n_start + NTOK / NCHUNK; n0 += 32) {
        const int np0 = n0 >> 1;
        // prefetch next tiles (lowers to global_prefetch_b8)
        __builtin_prefetch(ep + np0 + 16, 0, 1);
        __builtin_prefetch(&xp[(size_t)(dh * 256 + l15) * NPAIR + np0 + 16], 0, 1);

        ABfrag a;
        a.q[0] = *(const v4u*)(ep + np0 + hi * 8);
        a.q[1] = *(const v4u*)(ep + np0 + hi * 8 + 4);

        #pragma unroll
        for (int t = 0; t < 16; ++t) {
            int d = dh * 256 + t * 16 + l15;
            ABfrag b;
            if (PACKED) {
                const unsigned int* bb = xp + (size_t)d * NPAIR + np0 + hi * 8;
                b.q[0] = *(const v4u*)(bb);
                b.q[1] = *(const v4u*)(bb + 4);
            } else {
                #pragma unroll
                for (int v = 0; v < 8; ++v) {
                    int nb = n0 + 2 * v + hi * 16;
                    b.u[v] = pack2bf(x[(size_t)nb * DIM + d],
                                     x[(size_t)(nb + 1) * DIM + d]);
                }
            }
            acc[t] = __builtin_amdgcn_wmma_f32_16x16x32_bf16(
                false, a.v, false, b.v, (short)0, acc[t], false, false);
        }
    }

    #pragma unroll
    for (int t = 0; t < 16; ++t) {
        int d = dh * 256 + t * 16 + l15;
        #pragma unroll
        for (int r = 0; r < 8; ++r) {
            int krow = m4 * 16 + r + 8 * hi;
            atomicAdd(&ws[WS_ACCUM + (size_t)krow * DIM + d], acc[t][r]);
        }
    }
}

// ---------------------------------------------------------------------------
// Kernel 6: normalize
// ---------------------------------------------------------------------------
__global__ void finalize_kernel(const float* __restrict__ ws, float* __restrict__ out)
{
    int i = blockIdx.x * blockDim.x + threadIdx.x;
    if (i < KTOK * DIM) {
        int k = i / DIM;
        out[i] = ws[WS_ACCUM + i] / ws[WS_SUMEXP + k];
    }
}

// ---------------------------------------------------------------------------
extern "C" void kernel_launch(void* const* d_in, const int* in_sizes, int n_in,
                              void* d_out, int out_size, void* d_ws, size_t ws_size,
                              hipStream_t stream)
{
    const float* x     = (const float*)d_in[0];
    const float* gamma = (const float*)d_in[1];
    const float* beta  = (const float*)d_in[2];
    const float* W1    = (const float*)d_in[3];
    const float* b1    = (const float*)d_in[4];
    const float* W2    = (const float*)d_in[5];
    const float* b2    = (const float*)d_in[6];
    const float* scale = (const float*)d_in[7];
    float* ws  = (float*)d_ws;
    float* out = (float*)d_out;

    const bool packed = ws_size >= (size_t)WS_END_XP * sizeof(float);

    pack_weights_kernel<<<(HPAD * (DIM / 2) + KTOK * (HPAD / 2) + 255) / 256, 256, 0, stream>>>(W1, W2, ws);
    if (packed)
        pack_x_kernel<<<(NTOK / 64) * (DIM / 64), 256, 0, stream>>>(x, ws);
    score_kernel<<<NBLK1, 256, 0, stream>>>(x, gamma, beta, b1, b2, scale, ws);
    reduce_init_kernel<<<1, 256, 0, stream>>>(ws);
    epack_kernel<<<NEBLK, 256, 0, stream>>>(ws);
    reduce_sum_kernel<<<1, 64, 0, stream>>>(ws);
    if (packed)
        agg_kernel<true><<<NCHUNK, 256, 0, stream>>>(x, ws);
    else
        agg_kernel<false><<<NCHUNK, 256, 0, stream>>>(x, ws);
    finalize_kernel<<<(KTOK * DIM + 255) / 256, 256, 0, stream>>>(ws, out);
}
